// VGAE_Encoder_30700426232143
// MI455X (gfx1250) — matrix operand order repaired and checked by reading, not compiled, not run on hardware
//
#include <hip/hip_runtime.h>
#include <hip/hip_bf16.h>

typedef __attribute__((ext_vector_type(16))) __bf16 v16bf;
typedef __attribute__((ext_vector_type(8)))  float  v8f;
typedef __attribute__((ext_vector_type(4)))  int    v4i;

#define GLOBAL_AS __attribute__((address_space(1)))
#define LDS_AS    __attribute__((address_space(3)))

#if __has_builtin(__builtin_amdgcn_global_load_async_to_lds_b128) && \
    __has_builtin(__builtin_amdgcn_s_wait_asynccnt)
#define USE_ASYNC_LDS 1
#else
#define USE_ASYNC_LDS 0
#endif

// ---------------- degree / norm ----------------

__global__ void deg_init_k(float* __restrict__ deg, int N) {
    int i = blockIdx.x * blockDim.x + threadIdx.x;
    if (i < N) deg[i] = 1.0f;                // self-loop contributes 1
}

__global__ void deg_accum_k(const long long* __restrict__ ei, float* __restrict__ deg, int E) {
    int e = blockIdx.x * blockDim.x + threadIdx.x;
    if (e >= E) return;
    long long d = ei[(size_t)E + e];         // dst row of edge_index
    unsafeAtomicAdd(&deg[d], 1.0f);          // native global_atomic_add_f32
}

__global__ void deg_to_dinv_k(float* __restrict__ deg, int N) {
    int i = blockIdx.x * blockDim.x + threadIdx.x;
    if (i < N) deg[i] = rsqrtf(deg[i]);      // deg >= 1 always (self-loop)
}

// ---------------- weight transpose to bf16 (one-time, tiny) ----------------
// W: [128][Ncols] f32 row-major  ->  Wt: [Ncols][128] bf16 (K contiguous per column)
__global__ void transpose_w_bf16_k(const float* __restrict__ W, __bf16* __restrict__ Wt,
                                   int Ncols) {
    int idx = blockIdx.x * blockDim.x + threadIdx.x;
    if (idx >= 128 * Ncols) return;
    int k = idx / Ncols, n = idx % Ncols;
    Wt[(size_t)n * 128 + k] = (__bf16)W[idx];
}

// ---------------- aggregation (aggregate-first GCN) ----------------

// agg[i][:] = feat[i][:] * dinv[i]^2   (self-loop term; also zero-initializes agg)
__global__ void scaled_copy_k(const float* __restrict__ feat, const float* __restrict__ dinv,
                              float* __restrict__ agg, int N) {
    int gid  = blockIdx.x * blockDim.x + threadIdx.x;
    int i    = gid >> 5;
    int lane = gid & 31;
    if (i >= N) return;
    float w = dinv[i]; w *= w;
    const float4 v = *(const float4*)(feat + (size_t)i * 128 + lane * 4);
    float4 o; o.x = v.x * w; o.y = v.y * w; o.z = v.z * w; o.w = v.w * w;
    *(float4*)(agg + (size_t)i * 128 + lane * 4) = o;
}

// one wave (32 lanes) per edge: agg[dst] += feat[src] * dinv[src]*dinv[dst]
__global__ void edge_scatter_k(const long long* __restrict__ ei,
                               const float* __restrict__ feat,
                               const float* __restrict__ dinv,
                               float* __restrict__ agg, int E) {
    int gid  = blockIdx.x * blockDim.x + threadIdx.x;
    int e    = gid >> 5;
    int lane = gid & 31;
    if (e >= E) return;
    long long s = ei[e];
    long long d = ei[(size_t)E + e];
    float w = dinv[s] * dinv[d];
    const float4 v = *(const float4*)(feat + (size_t)s * 128 + lane * 4);
    float* dp = agg + (size_t)d * 128 + lane * 4;
    unsafeAtomicAdd(dp + 0, v.x * w);
    unsafeAtomicAdd(dp + 1, v.y * w);
    unsafeAtomicAdd(dp + 2, v.z * w);
    unsafeAtomicAdd(dp + 3, v.w * w);
}

// ---------------- WMMA GEMM helpers ----------------

// Stage the block's 16x128 f32 A-tile (contiguous 8KB) into LDS.
__device__ __forceinline__ void stage_A_tile(const float* __restrict__ A, int m0, float* As) {
#if USE_ASYNC_LDS
#pragma unroll
    for (int p = 0; p < 2; ++p) {
        int idx4 = (threadIdx.x + p * 256) * 4;   // float4 granules, 512 total
        __builtin_amdgcn_global_load_async_to_lds_b128(
            (GLOBAL_AS v4i*)(A + (size_t)m0 * 128 + idx4),
            (LDS_AS v4i*)(As + idx4), 0, 0);
    }
    __builtin_amdgcn_s_wait_asynccnt(0);
    __syncthreads();
#else
#pragma unroll
    for (int p = 0; p < 2; ++p) {
        int idx4 = (threadIdx.x + p * 256) * 4;
        *(float4*)(As + idx4) = *(const float4*)(A + (size_t)m0 * 128 + idx4);
    }
    __syncthreads();
#endif
}

// Build the A-fragment (16x32 bf16, contiguous k = half*16 + i order) from the LDS f32 tile.
__device__ __forceinline__ v16bf a_frag(const float* As, int mrow, int half, int k0) {
    v16bf a;
    const float* arow = As + mrow * 128 + k0 + half * 16;
#pragma unroll
    for (int i = 0; i < 8; ++i) {
        a[i]     = (__bf16)arow[i];          // K = k0 + half*16 + i
        a[8 + i] = (__bf16)arow[8 + i];      // K = k0 + half*16 + 8 + i
    }
    return a;
}

// ---------------- WMMA GEMMs ----------------
// Block: 16 output rows, 8 waves, A tile shared via LDS; per-wave 16x16 tile,
// K fully unrolled (4 x v_wmma_f32_16x16x32_bf16).

__global__ void __launch_bounds__(256)
gemm_relu_k(const float* __restrict__ A,        // [N,128] aggregated features
            const __bf16* __restrict__ Wt,      // [128 cols][128 K] bf16 transposed
            const float* __restrict__ bias,     // [128]
            float* __restrict__ H, int N) {
    __shared__ float As[16 * 128];
    int m0 = blockIdx.x * 16;
    stage_A_tile(A, m0, As);

    int wave = threadIdx.x >> 5;     // column tile 0..7
    int lane = threadIdx.x & 31;
    int mrow = lane & 15;
    int half = lane >> 4;
    int col  = wave * 16 + mrow;

    v8f c = {};
#pragma unroll
    for (int k0 = 0; k0 < 128; k0 += 32) {
        v16bf a = a_frag(As, mrow, half, k0);
        // B fragment: 16 consecutive K for this column -> one 32B contiguous load
        v16bf b = *(const v16bf*)(Wt + (size_t)col * 128 + k0 + half * 16);
        c = __builtin_amdgcn_wmma_f32_16x16x32_bf16(false, a, false, b,
                                                    (short)0, c, false, false);
    }
    float bv = bias[col];
#pragma unroll
    for (int r = 0; r < 8; ++r) {                // C/D: VGPR r -> M = r (+8 for hi lanes)
        int row = m0 + r + half * 8;
        H[(size_t)row * 128 + col] = fmaxf(c[r] + bv, 0.0f);
    }
}

// mu = A@Wmu + bmu ; logstd = A@Wls + bls ; concatenated into out.
__global__ void __launch_bounds__(256)
gemm_dual_k(const float* __restrict__ A,          // [N,128] aggregated hidden
            const __bf16* __restrict__ Wtmu, const float* __restrict__ bmu,
            const __bf16* __restrict__ Wtls, const float* __restrict__ bls,
            float* __restrict__ out, int N) {
    __shared__ float As[16 * 128];
    int m0 = blockIdx.x * 16;
    stage_A_tile(A, m0, As);

    int wave = threadIdx.x >> 5;                   // waves 0..3 -> mu, 4..7 -> logstd
    int lane = threadIdx.x & 31;
    const __bf16* Wt   = (wave < 4) ? Wtmu : Wtls; // wave-uniform; EXEC stays full
    const float*  bias = (wave < 4) ? bmu  : bls;
    float* dst = out + ((wave < 4) ? (size_t)0 : (size_t)N * 64);
    int mrow = lane & 15;
    int half = lane >> 4;
    int col  = (wave & 3) * 16 + mrow;             // 0..63 within this head

    v8f c = {};
#pragma unroll
    for (int k0 = 0; k0 < 128; k0 += 32) {
        v16bf a = a_frag(As, mrow, half, k0);
        v16bf b = *(const v16bf*)(Wt + (size_t)col * 128 + k0 + half * 16);
        c = __builtin_amdgcn_wmma_f32_16x16x32_bf16(false, a, false, b,
                                                    (short)0, c, false, false);
    }
    float bv = bias[col];
#pragma unroll
    for (int r = 0; r < 8; ++r) {
        int row = m0 + r + half * 8;
        dst[(size_t)row * 64 + col] = c[r] + bv;
    }
}

// ---------------- launch ----------------

extern "C" void kernel_launch(void* const* d_in, const int* in_sizes, int n_in,
                              void* d_out, int out_size, void* d_ws, size_t ws_size,
                              hipStream_t stream) {
    const float*     x   = (const float*)d_in[0];
    const long long* ei  = (const long long*)d_in[1];   // int64 [2,E]
    const float*     W1  = (const float*)d_in[2];
    const float*     b1  = (const float*)d_in[3];
    const float*     Wmu = (const float*)d_in[4];
    const float*     bmu = (const float*)d_in[5];
    const float*     Wls = (const float*)d_in[6];
    const float*     bls = (const float*)d_in[7];
    float* out = (float*)d_out;

    const int N = in_sizes[0] / 128;   // 100000 (multiple of 16)
    const int E = in_sizes[1] / 2;     // 1600000

    // workspace: dinv[N] | bufA[N*128] | bufB[N*128] | Wt1 | Wtmu | Wtls
    float*  wsf  = (float*)d_ws;
    size_t  npad = ((size_t)N + 1023) & ~((size_t)1023);
    float*  dinv = wsf;
    float*  bufA = wsf + npad;
    float*  bufB = bufA + (size_t)N * 128;
    __bf16* Wt1  = (__bf16*)(bufB + (size_t)N * 128);
    __bf16* Wtmu = Wt1 + 128 * 128;
    __bf16* Wtls = Wtmu + 64 * 128;

    const int TPB = 256;
    int nb_N   = (N + TPB - 1) / TPB;
    int nb_E   = (E + TPB - 1) / TPB;
    int nb_N32 = (int)(((size_t)N * 32 + TPB - 1) / TPB);
    int nb_E32 = (int)(((size_t)E * 32 + TPB - 1) / TPB);
    int nb_G   = N / 16;

    // 0) one-time transposed bf16 weights (K contiguous per output column)
    transpose_w_bf16_k<<<(128 * 128 + TPB - 1) / TPB, TPB, 0, stream>>>(W1,  Wt1,  128);
    transpose_w_bf16_k<<<(128 * 64  + TPB - 1) / TPB, TPB, 0, stream>>>(Wmu, Wtmu, 64);
    transpose_w_bf16_k<<<(128 * 64  + TPB - 1) / TPB, TPB, 0, stream>>>(Wls, Wtls, 64);

    // 1) symmetric degree norm: dinv = rsqrt(deg + 1)
    deg_init_k<<<nb_N, TPB, 0, stream>>>(dinv, N);
    deg_accum_k<<<nb_E, TPB, 0, stream>>>(ei, dinv, E);
    deg_to_dinv_k<<<nb_N, TPB, 0, stream>>>(dinv, N);

    // 2) layer 1: aggregate raw x, then dense transform + ReLU (aggregate-first GCN)
    scaled_copy_k<<<nb_N32, TPB, 0, stream>>>(x, dinv, bufA, N);       // self-loops
    edge_scatter_k<<<nb_E32, TPB, 0, stream>>>(ei, x, dinv, bufA, E);  // edges
    gemm_relu_k<<<nb_G, TPB, 0, stream>>>(bufA, Wt1, b1, bufB, N);     // h in bufB

    // 3) layer 2: aggregate h once, both heads share the aggregate
    scaled_copy_k<<<nb_N32, TPB, 0, stream>>>(bufB, dinv, bufA, N);
    edge_scatter_k<<<nb_E32, TPB, 0, stream>>>(ei, bufB, dinv, bufA, E);
    gemm_dual_k<<<nb_G, TPB, 0, stream>>>(bufA, Wtmu, bmu, Wtls, bls, out, N);
}
